// SlidingWindowAttention_19756849562258
// MI455X (gfx1250) — matrix operand order repaired and checked by reading, not compiled, not run on hardware
//
#include <hip/hip_runtime.h>
#include <hip/hip_bf16.h>

// ---------------------------------------------------------------------------
// Sliding-window attention (inverted-mask variant) for MI455X / gfx1250.
// B=2, S=2048, E=1024, H=16, D=64, WINDOW=3.
//   1) qkv_gemm<layout> (x3): Out = X @ W^T + b -> bf16 workspace.
//      Q,K stored [BH, S, D]; V stored transposed [BH, D, S].
//   2) attn: flash-style online-softmax attention, v_wmma_f32_16x16x32_bf16
//      for Q*K^T and P*V; K/V tiles staged via global_load_async_to_lds_b128
//      (ASYNCcnt) instead of VGPR round-trips.
// ---------------------------------------------------------------------------

#define N_B     2
#define N_S     2048
#define N_E     1024
#define N_H     16
#define N_D     64
#define N_BH    (N_B * N_H)
#define N_M     (N_B * N_S)
#define WINDOW  3

typedef __attribute__((ext_vector_type(16))) __bf16 v16bf;
typedef __attribute__((ext_vector_type(8)))  float  v8f;

union Frag {
    v16bf v;
    uint4 q[2];
};

static __device__ __forceinline__ unsigned short f32_to_bf16(float f) {
    unsigned int u = __builtin_bit_cast(unsigned int, f);
    u += 0x7FFFu + ((u >> 16) & 1u);          // round-to-nearest-even
    return (unsigned short)(u >> 16);
}

static __device__ __forceinline__ v8f vzero8() {
    v8f z;
#pragma unroll
    for (int i = 0; i < 8; ++i) z[i] = 0.0f;
    return z;
}

static __device__ __forceinline__ v8f wmma_bf16(v16bf a, v16bf b, v8f c) {
    return __builtin_amdgcn_wmma_f32_16x16x32_bf16(
        false, a, false, b, (short)0, c, false, false);
}

// Low 32 bits of a generic pointer into LDS == wave-relative LDS byte offset
// (ISA: shared aperture lives in ADDR[63:32]; ADDR[31:0] is the LDS address).
static __device__ __forceinline__ unsigned lds_off(const void* p) {
    return (unsigned)(unsigned long long)(uintptr_t)p;
}

// Async DMA: 16 bytes per lane, global -> LDS, tracked by ASYNCcnt.
static __device__ __forceinline__ void async_copy_b128(unsigned lds_addr,
                                                       unsigned long long gaddr) {
    asm volatile("global_load_async_to_lds_b128 %0, %1, off"
                 :: "v"(lds_addr), "v"(gaddr) : "memory");
}

static __device__ __forceinline__ void wait_async0() {
    asm volatile("s_wait_asynccnt 0" ::: "memory");
}

// ---------------------------------------------------------------------------
// Kernel 1: Out = X @ W^T + bias, bf16 output.
//   VLAYOUT == 0 : out[((b*16+h)*2048 + s)*64 + d]   (Q, K)
//   VLAYOUT == 1 : out[((b*16+h)*64 + d)*2048 + s]   (V, pre-transposed)
// Block: 256 threads (8 wave32), tile 128x128, K step 32.
// ---------------------------------------------------------------------------
template <int VLAYOUT>
__global__ __launch_bounds__(256) void qkv_gemm_kernel(
    const float* __restrict__ X,
    const float* __restrict__ W,
    const float* __restrict__ bias,
    unsigned short* __restrict__ out)
{
    __shared__ __align__(16) unsigned short At[128][32];  // [m][k] bf16
    __shared__ __align__(16) unsigned short Bt[128][32];  // [n][k] bf16

    const int tid  = threadIdx.x;
    const int lane = tid & 31;
    const int wv   = tid >> 5;
    const int wm   = (wv & 3) * 32;
    const int wn   = (wv >> 2) * 64;
    const int half = lane >> 4;
    const int l15  = lane & 15;

    const int mbase = blockIdx.y * 128;
    const int nbase = blockIdx.x * 128;

    v8f acc[2][4];
#pragma unroll
    for (int mi = 0; mi < 2; ++mi)
#pragma unroll
        for (int ni = 0; ni < 4; ++ni) acc[mi][ni] = vzero8();

    for (int kb = 0; kb < N_E; kb += 32) {
        // ---- stage fp32 -> bf16 tiles into LDS (conversion => VGPR path) ----
#pragma unroll
        for (int i = 0; i < 4; ++i) {
            const int idx = tid + i * 256;
            const int r   = idx >> 3;
            const int c4  = (idx & 7) * 4;
            const float4 xa = *reinterpret_cast<const float4*>(
                &X[(size_t)(mbase + r) * N_E + kb + c4]);
            ushort4 pa;
            pa.x = f32_to_bf16(xa.x); pa.y = f32_to_bf16(xa.y);
            pa.z = f32_to_bf16(xa.z); pa.w = f32_to_bf16(xa.w);
            *reinterpret_cast<ushort4*>(&At[r][c4]) = pa;

            const float4 wa = *reinterpret_cast<const float4*>(
                &W[(size_t)(nbase + r) * N_E + kb + c4]);
            ushort4 pb;
            pb.x = f32_to_bf16(wa.x); pb.y = f32_to_bf16(wa.y);
            pb.z = f32_to_bf16(wa.z); pb.w = f32_to_bf16(wa.w);
            *reinterpret_cast<ushort4*>(&Bt[r][c4]) = pb;
        }
        __syncthreads();

        Frag a[2], b[4];
#pragma unroll
        for (int mi = 0; mi < 2; ++mi) {
            const int row = wm + mi * 16 + l15;
            const int k0  = half * 8;
            a[mi].q[0] = *reinterpret_cast<const uint4*>(&At[row][k0]);
            a[mi].q[1] = *reinterpret_cast<const uint4*>(&At[row][k0 + 16]);
        }
#pragma unroll
        for (int ni = 0; ni < 4; ++ni) {
            const int col = wn + ni * 16 + l15;
            const int k0  = half * 8;
            b[ni].q[0] = *reinterpret_cast<const uint4*>(&Bt[col][k0]);
            b[ni].q[1] = *reinterpret_cast<const uint4*>(&Bt[col][k0 + 16]);
        }

#pragma unroll
        for (int mi = 0; mi < 2; ++mi)
#pragma unroll
            for (int ni = 0; ni < 4; ++ni)
                acc[mi][ni] = wmma_bf16(a[mi].v, b[ni].v, acc[mi][ni]);

        __syncthreads();
    }

    // ---- epilogue: + bias, bf16, store (layout is compile-time) ----
#pragma unroll
    for (int ni = 0; ni < 4; ++ni) {
        const int   n  = nbase + wn + ni * 16 + l15;
        const float bn = bias[n];
        const int   h  = n >> 6;
        const int   d  = n & 63;
#pragma unroll
        for (int mi = 0; mi < 2; ++mi) {
            const int m0 = mbase + wm + mi * 16 + half * 8;
            const int bI = m0 >> 11;
            const int s0 = m0 & 2047;
            if (VLAYOUT) {
                // [BH, D, S]: consecutive i are contiguous in s
                unsigned short* p =
                    out + ((((size_t)bI * N_H + h) * N_D + d) * N_S + s0);
#pragma unroll
                for (int i = 0; i < 8; ++i)
                    p[i] = f32_to_bf16(acc[mi][ni][i] + bn);
            } else {
                // [BH, S, D]: consecutive i stride by 64
                unsigned short* p =
                    out + ((((size_t)bI * N_H + h) * N_S + s0) * N_D + d);
#pragma unroll
                for (int i = 0; i < 8; ++i)
                    p[(size_t)i * N_D] = f32_to_bf16(acc[mi][ni][i] + bn);
            }
        }
    }
}

// ---------------------------------------------------------------------------
// Kernel 2: streaming-softmax attention.
//   Q,K: [BH, S, D] bf16 ; V: [BH, D, S] bf16 ; out: [B, S, E] f32.
//   grid = (S/64, BH); block = 128 threads (4 wave32), 16 query rows/wave.
// ---------------------------------------------------------------------------
__global__ __launch_bounds__(128) void attn_kernel(
    const unsigned short* __restrict__ Q,
    const unsigned short* __restrict__ K,
    const unsigned short* __restrict__ V,
    float* __restrict__ out)
{
    __shared__ __align__(16) unsigned short Kt[64][64];      // [key][d]
    __shared__ __align__(16) unsigned short Vt[64][64];      // [d][key]
    __shared__ __align__(16) unsigned short Pb[4][16][64];   // per-wave P

    const int tid  = threadIdx.x;
    const int lane = tid & 31;
    const int wv   = tid >> 5;
    const int half = lane >> 4;
    const int l15  = lane & 15;

    const int bh = blockIdx.y;
    const int qb = blockIdx.x * 64;
    const size_t qkBase = (size_t)bh * N_S * N_D;
    const size_t vBase  = (size_t)bh * N_D * N_S;

    const unsigned ktL = lds_off(&Kt[0][0]);
    const unsigned vtL = lds_off(&Vt[0][0]);

    // ---- this wave's Q fragments (16 rows x 64 d), kept in VGPRs ----
    Frag qa[2];
    {
        const int row = qb + wv * 16 + l15;
        const unsigned short* qp = Q + qkBase + (size_t)row * N_D;
        const int k0 = half * 8;
        qa[0].q[0] = *reinterpret_cast<const uint4*>(qp + k0);
        qa[0].q[1] = *reinterpret_cast<const uint4*>(qp + k0 + 16);
        qa[1].q[0] = *reinterpret_cast<const uint4*>(qp + 32 + k0);
        qa[1].q[1] = *reinterpret_cast<const uint4*>(qp + 32 + k0 + 16);
    }

    v8f o[4];
#pragma unroll
    for (int nt = 0; nt < 4; ++nt) o[nt] = vzero8();
    float mrow[8], lrow[8];
#pragma unroll
    for (int i = 0; i < 8; ++i) { mrow[i] = -1.0e30f; lrow[i] = 0.0f; }

    const float scale = 0.125f;          // 1/sqrt(64)

    for (int kb = 0; kb < N_S; kb += 64) {
        if (kb + 64 < N_S) {
            __builtin_prefetch(K + qkBase + (size_t)(kb + 64) * N_D + tid * 32, 0, 1);
            __builtin_prefetch(V + vBase + (size_t)(tid & 63) * N_S + kb + 64, 0, 1);
        }

        // ---- async DMA: K tile (contiguous) and V tile (row-strided) ----
        {
            const unsigned long long kg =
                (unsigned long long)(uintptr_t)(K + qkBase + (size_t)kb * N_D);
            const unsigned long long vg =
                (unsigned long long)(uintptr_t)(V + vBase + kb);
#pragma unroll
            for (int i = 0; i < 4; ++i) {
                const int slot = tid + i * 128;          // 512 x 16B = 8 KB
                async_copy_b128(ktL + (unsigned)slot * 16u,
                                kg + (unsigned long long)slot * 16ull);
                const int r = slot >> 3;                 // d row
                const int c = (slot & 7) * 8;            // key col (halves)
                async_copy_b128(vtL + ((unsigned)r * 64u + (unsigned)c) * 2u,
                                vg + ((unsigned long long)r * N_S + c) * 2ull);
            }
            wait_async0();                               // ASYNCcnt -> 0
        }
        __syncthreads();

        // ---- S = (Q K^T) * scale, inverted-window mask ----
        v8f s[4];
#pragma unroll
        for (int nt = 0; nt < 4; ++nt) {
            s[nt] = vzero8();
            const int col = nt * 16 + l15;
            Frag b0, b1;
            b0.q[0] = *reinterpret_cast<const uint4*>(&Kt[col][half * 8]);
            b0.q[1] = *reinterpret_cast<const uint4*>(&Kt[col][half * 8 + 16]);
            b1.q[0] = *reinterpret_cast<const uint4*>(&Kt[col][32 + half * 8]);
            b1.q[1] = *reinterpret_cast<const uint4*>(&Kt[col][32 + half * 8 + 16]);
            s[nt] = wmma_bf16(qa[0].v, b0.v, s[nt]);
            s[nt] = wmma_bf16(qa[1].v, b1.v, s[nt]);
        }

        float tmax[8];
#pragma unroll
        for (int i = 0; i < 8; ++i) tmax[i] = -1.0e30f;
#pragma unroll
        for (int nt = 0; nt < 4; ++nt) {
            const int kcol = kb + nt * 16 + l15;
#pragma unroll
            for (int i = 0; i < 8; ++i) {
                float v = s[nt][i] * scale;
                const int qrow = qb + wv * 16 + half * 8 + i;
                const int dlt  = qrow - kcol;
                if (dlt >= -WINDOW && dlt <= WINDOW) v = -1.0e30f;  // inverted mask
                s[nt][i] = v;
                tmax[i]  = fmaxf(tmax[i], v);
            }
        }
#pragma unroll
        for (int off = 1; off < 16; off <<= 1)
#pragma unroll
            for (int i = 0; i < 8; ++i)
                tmax[i] = fmaxf(tmax[i], __shfl_xor(tmax[i], off));

        float alpha[8];
#pragma unroll
        for (int i = 0; i < 8; ++i) {
            const float mn = fmaxf(mrow[i], tmax[i]);
            alpha[i] = __expf(mrow[i] - mn);
            mrow[i]  = mn;
            lrow[i] *= alpha[i];
        }
#pragma unroll
        for (int nt = 0; nt < 4; ++nt)
#pragma unroll
            for (int i = 0; i < 8; ++i) o[nt][i] *= alpha[i];

        float rsum[8];
#pragma unroll
        for (int i = 0; i < 8; ++i) rsum[i] = 0.0f;
#pragma unroll
        for (int nt = 0; nt < 4; ++nt) {
#pragma unroll
            for (int i = 0; i < 8; ++i) {
                const float p = __expf(s[nt][i] - mrow[i]);
                rsum[i] += p;
                Pb[wv][half * 8 + i][nt * 16 + l15] = f32_to_bf16(p);
            }
        }
#pragma unroll
        for (int off = 1; off < 16; off <<= 1)
#pragma unroll
            for (int i = 0; i < 8; ++i)
                rsum[i] += __shfl_xor(rsum[i], off);
#pragma unroll
        for (int i = 0; i < 8; ++i) lrow[i] += rsum[i];

        __syncthreads();   // order Pb stores before A-frag reads

        // ---- O += P @ V ----
        Frag pa[2];
#pragma unroll
        for (int ks = 0; ks < 2; ++ks) {
            const int k0 = ks * 32 + half * 8;
            pa[ks].q[0] = *reinterpret_cast<const uint4*>(&Pb[wv][l15][k0]);
            pa[ks].q[1] = *reinterpret_cast<const uint4*>(&Pb[wv][l15][k0 + 16]);
        }
#pragma unroll
        for (int nt = 0; nt < 4; ++nt) {
            const int col = nt * 16 + l15;           // d
            Frag v0, v1;
            v0.q[0] = *reinterpret_cast<const uint4*>(&Vt[col][half * 8]);
            v0.q[1] = *reinterpret_cast<const uint4*>(&Vt[col][half * 8 + 16]);
            v1.q[0] = *reinterpret_cast<const uint4*>(&Vt[col][32 + half * 8]);
            v1.q[1] = *reinterpret_cast<const uint4*>(&Vt[col][32 + half * 8 + 16]);
            o[nt] = wmma_bf16(pa[0].v, v0.v, o[nt]);
            o[nt] = wmma_bf16(pa[1].v, v1.v, o[nt]);
        }
        __syncthreads();   // all waves done reading before next async stage
    }

    // ---- normalize, store fp32 [B, S, E] ----
    const int bI = bh >> 4;
    const int h  = bh & 15;
    float inv[8];
#pragma unroll
    for (int i = 0; i < 8; ++i) inv[i] = 1.0f / lrow[i];
#pragma unroll
    for (int nt = 0; nt < 4; ++nt) {
        const int e = h * N_D + nt * 16 + l15;
#pragma unroll
        for (int i = 0; i < 8; ++i) {
            const int s = qb + wv * 16 + half * 8 + i;
            out[((size_t)bI * N_S + s) * N_E + e] = o[nt][i] * inv[i];
        }
    }
}

// ---------------------------------------------------------------------------
extern "C" void kernel_launch(void* const* d_in, const int* in_sizes, int n_in,
                              void* d_out, int out_size, void* d_ws, size_t ws_size,
                              hipStream_t stream) {
    const float* X  = (const float*)d_in[0];
    const float* Wq = (const float*)d_in[1];
    const float* bq = (const float*)d_in[2];
    const float* Wk = (const float*)d_in[3];
    const float* bk = (const float*)d_in[4];
    const float* Wv = (const float*)d_in[5];
    const float* bv = (const float*)d_in[6];

    const size_t perMat = (size_t)N_BH * N_S * N_D;
    unsigned short* Qw = (unsigned short*)d_ws;
    unsigned short* Kw = Qw + perMat;
    unsigned short* Vw = Kw + perMat;

    dim3 gGemm(N_E / 128, N_M / 128);
    qkv_gemm_kernel<0><<<gGemm, 256, 0, stream>>>(X, Wq, bq, Qw);
    qkv_gemm_kernel<0><<<gGemm, 256, 0, stream>>>(X, Wk, bk, Kw);
    qkv_gemm_kernel<1><<<gGemm, 256, 0, stream>>>(X, Wv, bv, Vw);

    dim3 gAttn(N_S / 64, N_BH);
    attn_kernel<<<gAttn, 128, 0, stream>>>(Qw, Kw, Vw, (float*)d_out);
}
